// WBLMoE_11570641895579
// MI455X (gfx1250) — matrix operand order repaired and checked by reading, not compiled, not run on hardware
//
#include <hip/hip_runtime.h>
#include <math.h>

// ---------------- problem constants (match reference) ----------------
#define NTOK 2048
#define HDIM 1024
#define NEXP 16
#define TOPK 4
#define IDIM 512          // routed intermediate
#define ISH  1024         // shared intermediate (I*NS)
#define EI   (NEXP*IDIM)  // 8192 : big-K for fused routed down GEMM
#define ROUTED_SCALE 2.5f

// ---------------- WMMA types ----------------
typedef __attribute__((ext_vector_type(8)))  float  v8f;
typedef __attribute__((ext_vector_type(16))) __bf16 v16bf;
typedef __attribute__((ext_vector_type(2)))  float  v2f;
typedef __attribute__((ext_vector_type(2)))  __bf16 v2bf;

union Frag { uint4 u[2]; v16bf v; };

__device__ __forceinline__ unsigned short f32_to_bf16(float f) {
  union { __bf16 h; unsigned short s; } c;
  c.h = (__bf16)f;                     // fptrunc, RNE
  return c.s;
}
__device__ __forceinline__ unsigned int pack2(float a, float b) {
  v2f f = {a, b};
  v2bf h = __builtin_convertvector(f, v2bf);   // native pk-cvt if available
  union { v2bf v; unsigned int i; } c;
  c.v = h;
  return c.i;
}
__device__ __forceinline__ v8f zero8() {
  v8f z = {0.f,0.f,0.f,0.f,0.f,0.f,0.f,0.f};
  return z;
}

#define LST 40   // LDS row stride (halves): 80B rows -> conflict-free frag reads

// =====================================================================
// 1) Router: one wave (32 lanes) per token.
// =====================================================================
__global__ __launch_bounds__(256) void router_kernel(
    const float* __restrict__ x, const float* __restrict__ gate_w,
    float* __restrict__ w_dense)
{
  const int lane = threadIdx.x & 31;
  const int wave = threadIdx.x >> 5;
  const int n = blockIdx.x * 8 + wave;
  if (n >= NTOK) return;

  float acc[NEXP];
#pragma unroll
  for (int e = 0; e < NEXP; ++e) acc[e] = 0.f;

  const float* xr = x + (size_t)n * HDIM;
  for (int h = lane; h < HDIM; h += 32) {
    float xv = xr[h];
    const float* gw = gate_w + (size_t)h * NEXP;
#pragma unroll
    for (int e = 0; e < NEXP; ++e) acc[e] += xv * gw[e];
  }
#pragma unroll
  for (int off = 16; off > 0; off >>= 1) {
#pragma unroll
    for (int e = 0; e < NEXP; ++e) acc[e] += __shfl_xor(acc[e], off, 32);
  }

  if (lane == 0) {
    float s[NEXP];
#pragma unroll
    for (int e = 0; e < NEXP; ++e) s[e] = 1.f / (1.f + __expf(-acc[e]));
    int used[NEXP];
#pragma unroll
    for (int e = 0; e < NEXP; ++e) used[e] = 0;
    float tv[TOPK]; int ti[TOPK]; float tsum = 0.f;
#pragma unroll
    for (int j = 0; j < TOPK; ++j) {
      float best = -1.f; int bi = 0;
#pragma unroll
      for (int e = 0; e < NEXP; ++e)
        if (!used[e] && s[e] > best) { best = s[e]; bi = e; }
      used[bi] = 1; tv[j] = best; ti[j] = bi; tsum += best;
    }
    float* wd = w_dense + (size_t)n * NEXP;
#pragma unroll
    for (int e = 0; e < NEXP; ++e) wd[e] = 0.f;
#pragma unroll
    for (int j = 0; j < TOPK; ++j) wd[ti[j]] = tv[j] / tsum;
  }
}

// =====================================================================
// 2) x (f32) -> xb (bf16), 8 elements / thread
// =====================================================================
__global__ __launch_bounds__(256) void convert_x_kernel(
    const float* __restrict__ x, unsigned short* __restrict__ xb)
{
  const size_t g = (size_t)blockIdx.x * blockDim.x + threadIdx.x;
  const size_t base = g * 8;
  if (base >= (size_t)NTOK * HDIM) return;
  const float4 a = *(const float4*)(x + base);
  const float4 b = *(const float4*)(x + base + 4);
  uint4 q;
  q.x = pack2(a.x, a.y); q.y = pack2(a.z, a.w);
  q.z = pack2(b.x, b.y); q.w = pack2(b.z, b.w);
  *(uint4*)(xb + base) = q;
}

// =====================================================================
// 2b) One-time weight transform: f32 [R][C] -> bf16 transposed [C][R].
//     Makes GEMM B-staging a pure 16B copy (zero VALU in the hot loop)
//     and halves weight-fetch bytes there.
// =====================================================================
__global__ __launch_bounds__(256) void transpose_cvt_kernel(
    const float* __restrict__ in, unsigned short* __restrict__ out,
    int R, int C, long long inStrideZ, long long outStrideZ)
{
  __shared__ float tile[32][33];
  const float* inz = in + (size_t)blockIdx.z * inStrideZ;
  unsigned short* outz = out + (size_t)blockIdx.z * outStrideZ;
  const int r0 = blockIdx.x * 32, c0 = blockIdx.y * 32;
  const int tx = threadIdx.x, ty = threadIdx.y;
#pragma unroll
  for (int i = ty; i < 32; i += 8)
    tile[i][tx] = inz[(size_t)(r0 + i) * C + (c0 + tx)];
  __syncthreads();
#pragma unroll
  for (int i = ty; i < 32; i += 8)
    outz[(size_t)(c0 + i) * R + (r0 + tx)] = f32_to_bf16(tile[tx][i]);
}

// =====================================================================
// 3) SwiGLU gate_up GEMM. 64x64 tile / 256 threads (8 waves, each 16x32
//    of gate-C and up-C). Software-pipelined k-loop.
//    BF16W: B is pre-transposed bf16 [outCol][K] (pure-copy staging).
//    else : B is f32 [K][outCol], converted inline.
// =====================================================================
template <bool HAS_SCALE, bool BF16W>
__global__ __launch_bounds__(256) void gateup_kernel(
    const unsigned short* __restrict__ Abf,  // [M][lda] bf16
    const void* __restrict__ Bv,             // weights (see BF16W)
    unsigned short* __restrict__ Obf,        // bf16 out
    const float* __restrict__ scale,         // w_dense (per row,z); iff HAS_SCALE
    int lda, int Kdim, int ldb, int up_off, int ldo, int outColPerZ,
    long long bStrideZ)
{
  __shared__ __align__(16) unsigned short As[64 * LST];
  __shared__ __align__(16) unsigned short Bg[64 * LST];
  __shared__ __align__(16) unsigned short Bu[64 * LST];

  const int tid  = threadIdx.x;
  const int lane = tid & 31;
  const int wave = tid >> 5;
  const int wm   = wave & 3;     // 4 row strips of 16
  const int wn   = wave >> 2;    // 2 col strips of 32
  const int m0   = blockIdx.x * 64;
  const int f0   = blockIdx.y * 64;
  const int z    = blockIdx.z;
  const float*          Bzf = (const float*)Bv + (size_t)z * bStrideZ;
  const unsigned short* Bzh = (const unsigned short*)Bv + (size_t)z * bStrideZ;

  v8f cg[2], cu[2];
  cg[0] = cg[1] = cu[0] = cu[1] = zero8();

  const int arow = tid >> 2;      // 0..63
  const int akg  = tid & 3;       // 0..3 (8 halves each)
  const int bcol = tid & 63;      // 0..63
  const int bkg  = tid >> 6;      // 0..3 (8 k each)

  const int ll   = lane & 15;
  const int kbA  = (lane >> 4) * 8;
  const int kbB  = (lane >> 4) * 8;

  uint4 aReg, gH, uH;
  float gR[8], uR[8];

  auto loadA = [&](int k0) {
    aReg = *(const uint4*)(Abf + (size_t)(m0 + arow) * lda + k0 + akg * 8);
  };
  auto loadB = [&](int k0) {
    if constexpr (BF16W) {
      gH = *(const uint4*)(Bzh + (size_t)(f0 + bcol) * ldb + k0 + bkg * 8);
      uH = *(const uint4*)(Bzh + (size_t)(up_off + f0 + bcol) * ldb + k0 + bkg * 8);
    } else {
#pragma unroll
      for (int i = 0; i < 8; ++i) {
        size_t kk = (size_t)(k0 + bkg * 8 + i) * ldb;
        gR[i] = Bzf[kk + f0 + bcol];
        uR[i] = Bzf[kk + up_off + f0 + bcol];
      }
    }
  };

  loadA(0); loadB(0);

  for (int k0 = 0; k0 < Kdim; k0 += 32) {
    *(uint4*)(As + arow * LST + akg * 8) = aReg;
    if constexpr (BF16W) {
      *(uint4*)(Bg + bcol * LST + bkg * 8) = gH;
      *(uint4*)(Bu + bcol * LST + bkg * 8) = uH;
    } else {
      uint4 qg, qu;
      qg.x = pack2(gR[0], gR[1]); qg.y = pack2(gR[2], gR[3]);
      qg.z = pack2(gR[4], gR[5]); qg.w = pack2(gR[6], gR[7]);
      qu.x = pack2(uR[0], uR[1]); qu.y = pack2(uR[2], uR[3]);
      qu.z = pack2(uR[4], uR[5]); qu.w = pack2(uR[6], uR[7]);
      *(uint4*)(Bg + bcol * LST + bkg * 8) = qg;
      *(uint4*)(Bu + bcol * LST + bkg * 8) = qu;
    }
    __syncthreads();

    if (k0 + 32 < Kdim) { loadA(k0 + 32); loadB(k0 + 32); }  // overlap with WMMA

    Frag af;
    {
      const unsigned short* ap = As + (wm * 16 + ll) * LST;
      af.u[0] = *(const uint4*)(ap + kbA);
      af.u[1] = *(const uint4*)(ap + 16 + kbA);
    }
#pragma unroll
    for (int fi = 0; fi < 2; ++fi) {
      const int col = wn * 32 + fi * 16 + ll;
      Frag bg, bu;
      const unsigned short* bgp = Bg + col * LST + kbB;
      const unsigned short* bup = Bu + col * LST + kbB;
      bg.u[0] = *(const uint4*)(bgp);  bg.u[1] = *(const uint4*)(bgp + 16);
      bu.u[0] = *(const uint4*)(bup);  bu.u[1] = *(const uint4*)(bup + 16);
      cg[fi] = __builtin_amdgcn_wmma_f32_16x16x32_bf16(
          false, af.v, false, bg.v, (short)0, cg[fi], false, false);
      cu[fi] = __builtin_amdgcn_wmma_f32_16x16x32_bf16(
          false, af.v, false, bu.v, (short)0, cu[fi], false, false);
    }
    __syncthreads();
  }

  const int hi = (lane & 16) ? 8 : 0;
  float sc[8];
#pragma unroll
  for (int r = 0; r < 8; ++r) {
    if constexpr (HAS_SCALE) {
      const int m = m0 + wm * 16 + r + hi;
      sc[r] = scale[(size_t)m * NEXP + z];
    } else {
      sc[r] = 1.0f;
    }
  }
#pragma unroll
  for (int fi = 0; fi < 2; ++fi) {
    const int ocol = f0 + wn * 32 + fi * 16 + ll + z * outColPerZ;
#pragma unroll
    for (int r = 0; r < 8; ++r) {
      const int m = m0 + wm * 16 + r + hi;
      const float g = cg[fi][r];
      const float u = cu[fi][r];
      const float a = (g / (1.0f + __expf(-g))) * u * sc[r];
      Obf[(size_t)m * ldo + ocol] = f32_to_bf16(a);
    }
  }
}

// =====================================================================
// 4) Down GEMM: Out = alpha * (Abf @ B) [+ Out]; same pipelined scheme.
//    BF16W: B pre-transposed bf16 [n][K]; else f32 [K][n].
// =====================================================================
template <bool ADD, bool BF16W>
__global__ __launch_bounds__(256) void down_kernel(
    const unsigned short* __restrict__ Abf, int Kdim, int lda,
    const void* __restrict__ Bv, int ldb,
    float* __restrict__ Out, float alpha)
{
  __shared__ __align__(16) unsigned short As[64 * LST];
  __shared__ __align__(16) unsigned short Bt[64 * LST];

  const int tid  = threadIdx.x;
  const int lane = tid & 31;
  const int wave = tid >> 5;
  const int wm   = wave & 3;
  const int wn   = wave >> 2;
  const int m0   = blockIdx.x * 64;
  const int n0   = blockIdx.y * 64;
  const float*          Bf = (const float*)Bv;
  const unsigned short* Bh = (const unsigned short*)Bv;

  v8f c[2];
  c[0] = c[1] = zero8();

  const int arow = tid >> 2;
  const int akg  = tid & 3;
  const int bcol = tid & 63;
  const int bkg  = tid >> 6;

  const int ll  = lane & 15;
  const int kbA = (lane >> 4) * 8;
  const int kbB = (lane >> 4) * 8;

  uint4 aReg, bH;
  float bR[8];

  auto loadA = [&](int k0) {
    aReg = *(const uint4*)(Abf + (size_t)(m0 + arow) * lda + k0 + akg * 8);
  };
  auto loadB = [&](int k0) {
    if constexpr (BF16W) {
      bH = *(const uint4*)(Bh + (size_t)(n0 + bcol) * ldb + k0 + bkg * 8);
    } else {
#pragma unroll
      for (int i = 0; i < 8; ++i)
        bR[i] = Bf[(size_t)(k0 + bkg * 8 + i) * ldb + n0 + bcol];
    }
  };

  loadA(0); loadB(0);

  for (int k0 = 0; k0 < Kdim; k0 += 32) {
    *(uint4*)(As + arow * LST + akg * 8) = aReg;
    if constexpr (BF16W) {
      *(uint4*)(Bt + bcol * LST + bkg * 8) = bH;
    } else {
      uint4 q;
      q.x = pack2(bR[0], bR[1]); q.y = pack2(bR[2], bR[3]);
      q.z = pack2(bR[4], bR[5]); q.w = pack2(bR[6], bR[7]);
      *(uint4*)(Bt + bcol * LST + bkg * 8) = q;
    }
    __syncthreads();

    if (k0 + 32 < Kdim) { loadA(k0 + 32); loadB(k0 + 32); }

    Frag af;
    {
      const unsigned short* ap = As + (wm * 16 + ll) * LST;
      af.u[0] = *(const uint4*)(ap + kbA);
      af.u[1] = *(const uint4*)(ap + 16 + kbA);
    }
#pragma unroll
    for (int fi = 0; fi < 2; ++fi) {
      const int col = wn * 32 + fi * 16 + ll;
      Frag bfg;
      const unsigned short* bp = Bt + col * LST + kbB;
      bfg.u[0] = *(const uint4*)(bp);  bfg.u[1] = *(const uint4*)(bp + 16);
      c[fi] = __builtin_amdgcn_wmma_f32_16x16x32_bf16(
          false, af.v, false, bfg.v, (short)0, c[fi], false, false);
    }
    __syncthreads();
  }

  const int hi = (lane & 16) ? 8 : 0;
#pragma unroll
  for (int fi = 0; fi < 2; ++fi) {
    const int ncol = n0 + wn * 32 + fi * 16 + ll;
#pragma unroll
    for (int r = 0; r < 8; ++r) {
      const int m = m0 + wm * 16 + r + hi;
      float* p = Out + (size_t)m * HDIM + ncol;
      const float v = alpha * c[fi][r];
      if constexpr (ADD) *p = *p + v; else *p = v;
    }
  }
}

// =====================================================================
// Launch
// =====================================================================
extern "C" void kernel_launch(void* const* d_in, const int* in_sizes, int n_in,
                              void* d_out, int out_size, void* d_ws, size_t ws_size,
                              hipStream_t stream) {
  const float* x              = (const float*)d_in[0];
  const float* gate_w         = (const float*)d_in[1];
  const float* w_gate_up      = (const float*)d_in[2];
  const float* w_down         = (const float*)d_in[3];
  const float* shared_gate_up = (const float*)d_in[4];
  const float* shared_down    = (const float*)d_in[5];
  float* out = (float*)d_out;

  // ---- workspace layout ----
  const size_t SZ_WD   = (size_t)NTOK * NEXP * 4;        // 128 KB
  const size_t SZ_XB   = (size_t)NTOK * HDIM * 2;        // 4 MB
  const size_t SZ_ACT  = (size_t)NTOK * EI * 2;          // 32 MB
  const size_t SZ_HS   = (size_t)NTOK * ISH * 2;         // 4 MB
  const size_t SZ_WGUT = (size_t)NEXP * 2 * IDIM * HDIM * 2;  // 32 MB
  const size_t SZ_SGUT = (size_t)2 * ISH * HDIM * 2;          // 4 MB
  const size_t SZ_WDT  = (size_t)HDIM * EI * 2;               // 16 MB
  const size_t SZ_SDT  = (size_t)HDIM * ISH * 2;              // 2 MB

  char* ws = (char*)d_ws;
  size_t off = 0;
  float*          w_dense = (float*)(ws + off);          off += SZ_WD;
  unsigned short* xb      = (unsigned short*)(ws + off); off += SZ_XB;
  unsigned short* actb    = (unsigned short*)(ws + off); off += SZ_ACT;
  unsigned short* hs      = (unsigned short*)(ws + off); off += SZ_HS;
  const size_t small_need = off;
  unsigned short* wguT    = (unsigned short*)(ws + off); off += SZ_WGUT;
  unsigned short* sguT    = (unsigned short*)(ws + off); off += SZ_SGUT;
  unsigned short* wdT     = (unsigned short*)(ws + off); off += SZ_WDT;
  unsigned short* sdT     = (unsigned short*)(ws + off); off += SZ_SDT;
  const bool big = (ws_size >= off);
  (void)small_need;

  // 1) router + x->bf16
  router_kernel<<<NTOK / 8, 256, 0, stream>>>(x, gate_w, w_dense);
  convert_x_kernel<<<(NTOK * HDIM / 8) / 256, 256, 0, stream>>>(x, xb);

  if (big) {
    // one-time weight convert+transpose (bf16, K-contiguous)
    dim3 tb(32, 8, 1);
    transpose_cvt_kernel<<<dim3(HDIM/32, (2*IDIM)/32, NEXP), tb, 0, stream>>>(
        w_gate_up, wguT, HDIM, 2*IDIM,
        (long long)HDIM * 2 * IDIM, (long long)2 * IDIM * HDIM);
    transpose_cvt_kernel<<<dim3(HDIM/32, (2*ISH)/32, 1), tb, 0, stream>>>(
        shared_gate_up, sguT, HDIM, 2*ISH, 0LL, 0LL);
    transpose_cvt_kernel<<<dim3(EI/32, HDIM/32, 1), tb, 0, stream>>>(
        w_down, wdT, EI, HDIM, 0LL, 0LL);
    transpose_cvt_kernel<<<dim3(ISH/32, HDIM/32, 1), tb, 0, stream>>>(
        shared_down, sdT, ISH, HDIM, 0LL, 0LL);

    gateup_kernel<false, true><<<dim3(NTOK/64, ISH/64, 1), 256, 0, stream>>>(
        xb, sguT, hs, (const float*)nullptr,
        HDIM, HDIM, /*ldb=K*/ HDIM, /*up_off(row)*/ ISH, ISH, 0, 0LL);
    down_kernel<false, true><<<dim3(NTOK/64, HDIM/64, 1), 256, 0, stream>>>(
        hs, ISH, ISH, sdT, /*ldb=K*/ ISH, out, 1.0f);
    gateup_kernel<true, true><<<dim3(NTOK/64, IDIM/64, NEXP), 256, 0, stream>>>(
        xb, wguT, actb, w_dense,
        HDIM, HDIM, /*ldb=K*/ HDIM, /*up_off(row)*/ IDIM, EI, IDIM,
        (long long)2 * IDIM * HDIM);
    down_kernel<true, true><<<dim3(NTOK/64, HDIM/64, 1), 256, 0, stream>>>(
        actb, EI, EI, wdT, /*ldb=K*/ EI, out, ROUTED_SCALE);
  } else {
    // fallback: inline f32->bf16 conversion inside the GEMMs
    gateup_kernel<false, false><<<dim3(NTOK/64, ISH/64, 1), 256, 0, stream>>>(
        xb, shared_gate_up, hs, (const float*)nullptr,
        HDIM, HDIM, /*ldb*/ 2*ISH, /*up_off(col)*/ ISH, ISH, 0, 0LL);
    down_kernel<false, false><<<dim3(NTOK/64, HDIM/64, 1), 256, 0, stream>>>(
        hs, ISH, ISH, shared_down, /*ldb*/ HDIM, out, 1.0f);
    gateup_kernel<true, false><<<dim3(NTOK/64, IDIM/64, NEXP), 256, 0, stream>>>(
        xb, w_gate_up, actb, w_dense,
        HDIM, HDIM, /*ldb*/ 2*IDIM, /*up_off(col)*/ IDIM, EI, IDIM,
        (long long)HDIM * 2 * IDIM);
    down_kernel<true, false><<<dim3(NTOK/64, HDIM/64, 1), 256, 0, stream>>>(
        actb, EI, EI, w_down, /*ldb*/ HDIM, out, ROUTED_SCALE);
  }
}
